// GNN_DSR_64836826300739
// MI455X (gfx1250) — compile-verified, matrix operand order checked by reading, not run on hardware
//
#include <hip/hip_runtime.h>
#include <math.h>

#define D_  64
#define B_  256
#define U_  50
#define K_  50
#define U2_ 30

typedef __bf16 bf16_t;
typedef bf16_t bf16x16 __attribute__((ext_vector_type(16)));
typedef float  floatx8 __attribute__((ext_vector_type(8)));

// ---------------- bf16 weight arena (element offsets) ----------------
#define OFF_GU0    0         // 64x128
#define OFF_GU1    8192      // 64x64
#define OFF_W1     12288     // 64x64
#define OFF_W2     16384
#define OFF_W3     20480
#define OFF_ATTIU0 24576     // 64x128
#define OFF_ATTIF0 32768
#define OFF_ATTFF0 40960
#define OFF_LSTM   49152     // 4 layers x (Wih 256x64, Whh 256x64)
#define ARENA_ELEMS 180224
#define ARENA_BYTES ((size_t)ARENA_ELEMS * 2)

// ---------------- f32 workspace layout (float offsets, after arena) ----------------
static const size_t F_BIASC = 0;                         // 4*256
static const size_t F_Y1    = F_BIASC + 4*256;           // 12800*64
static const size_t F_W1Y   = F_Y1    + 12800ull*64;     // 12800*64
static const size_t F_LOG1  = F_W1Y   + 12800ull*64;     // 12800
static const size_t F_ZJL   = F_LOG1  + 12800;           // 256*64
static const size_t F_ZJS   = F_ZJL   + 256*64;          // 256*64
static const size_t F_LSEQ1 = F_ZJS   + 256*64;          // 12800*64
static const size_t F_YS    = F_LSEQ1 + 12800ull*64;     // 384000*64
static const size_t F_LOG2  = F_YS    + 384000ull*64;    // 384000
static const size_t F_ZKL   = F_LOG2  + 384000;          // 12800*64
static const size_t F_ZKS   = F_ZKL   + 12800ull*64;     // 12800*64
static const size_t F_ZKA   = F_ZKS   + 12800ull*64;     // 12800*64
static const size_t F_KLOG  = F_ZKA   + 12800ull*64;     // 12800
static const size_t F_ZJN   = F_KLOG  + 12800;           // 256*64
static const size_t F_LSEQ2 = F_ZJN   + 256*64;          // 384000*64

// =====================================================================
// WMMA helpers (CDNA5 wave32, v_wmma_f32_16x16x32_bf16)
// =====================================================================
__device__ __forceinline__ floatx8 zero8() {
  floatx8 z = {0.f,0.f,0.f,0.f,0.f,0.f,0.f,0.f};
  return z;
}

__device__ __forceinline__ floatx8 wmma_bf16(bf16x16 a, bf16x16 b, floatx8 c) {
  return __builtin_amdgcn_wmma_f32_16x16x32_bf16(false, a, false, b, (short)0, c, false, false);
}

// Pack one 16x32 bf16 fragment from a row-major LDS tile.
// Layout (ISA 7.12.2, 16-bit A 16x32; B packs identically with N<->M):
//  lanes 0-15 : row r=lane, elems 0-7 -> K = kBase+0..7,  elems 8-15 -> K = kBase+16..23
//  lanes 16-31: row r=lane-16, elems 0-7 -> K = kBase+8..15, elems 8-15 -> K = kBase+24..31
__device__ __forceinline__ bf16x16 pack_frag(const bf16_t* __restrict__ tile,
                                             int rowStride, int kBase, int lane) {
  const int r  = lane & 15;
  const int b1 = kBase + ((lane & 16) ? 8 : 0);
  const bf16_t* p = tile + r * rowStride;
  bf16x16 v;
#pragma unroll
  for (int j = 0; j < 8; ++j) {
    v[j]     = p[b1 + j];
    v[8 + j] = p[b1 + 16 + j];
  }
  return v;
}

// One wave computes [16 x KDIM] @ W[64 x KDIM]^T -> [16 x 64] (+bias, activation).
// act: 0=none, 1=leaky(0.2), 2=relu. Optional bf16 LDS output tile + f32 global rows.
template<int KDIM>
__device__ __forceinline__ void wave_gemm64(
    const bf16_t* __restrict__ xTile, int xStride,
    const bf16_t* __restrict__ wTile, int wStride,
    const float* __restrict__ bias, int act,
    bf16_t* outTile, int oStride,
    float* __restrict__ outF, int rowBase, int lane) {
  constexpr int KC = KDIM / 32;
  bf16x16 afr[KC];
#pragma unroll
  for (int kc = 0; kc < KC; ++kc) afr[kc] = pack_frag(xTile, xStride, kc * 32, lane);
  const int nsub = lane & 15;
  const int rsub = (lane & 16) ? 8 : 0;
#pragma unroll
  for (int nt = 0; nt < 4; ++nt) {
    floatx8 acc = zero8();
#pragma unroll
    for (int kc = 0; kc < KC; ++kc) {
      bf16x16 bfr = pack_frag(wTile + (nt * 16) * wStride, wStride, kc * 32, lane);
      acc = wmma_bf16(afr[kc], bfr, acc);
    }
    const int n = nt * 16 + nsub;
    const float bn = bias ? bias[n] : 0.f;
#pragma unroll
    for (int v = 0; v < 8; ++v) {
      float x = acc[v] + bn;
      if (act == 1)      x = (x > 0.f) ? x : 0.2f * x;
      else if (act == 2) x = (x > 0.f) ? x : 0.f;
      const int row = v + rsub;
      if (outTile) outTile[row * oStride + n] = (bf16_t)x;
      if (outF)    outF[(size_t)(rowBase + row) * 64 + n] = x;
    }
  }
}

// Attention tail: logit[row] = A1 . leaky(X[16x128] @ Att0[64x128]^T + b0) + a1b
__device__ __forceinline__ void wave_att_logit(
    const bf16_t* __restrict__ xTile, int xStride,
    const bf16_t* __restrict__ wTile, int wStride,
    const float* __restrict__ b0, const float* __restrict__ a1w,
    const float* __restrict__ a1b,
    float* __restrict__ outLogit, int rowBase, int lane) {
  bf16x16 afr[4];
#pragma unroll
  for (int kc = 0; kc < 4; ++kc) afr[kc] = pack_frag(xTile, xStride, kc * 32, lane);
  float part[8];
#pragma unroll
  for (int v = 0; v < 8; ++v) part[v] = 0.f;
  const int nsub = lane & 15;
#pragma unroll
  for (int nt = 0; nt < 4; ++nt) {
    floatx8 acc = zero8();
#pragma unroll
    for (int kc = 0; kc < 4; ++kc) {
      bf16x16 bfr = pack_frag(wTile + (nt * 16) * wStride, wStride, kc * 32, lane);
      acc = wmma_bf16(afr[kc], bfr, acc);
    }
    const int n = nt * 16 + nsub;
    const float bn = b0[n];
    const float a1 = a1w[n];
#pragma unroll
    for (int v = 0; v < 8; ++v) {
      float t = acc[v] + bn;
      t = (t > 0.f) ? t : 0.2f * t;
      part[v] += t * a1;
    }
  }
#pragma unroll
  for (int v = 0; v < 8; ++v) {
#pragma unroll
    for (int m = 1; m < 16; m <<= 1) part[v] += __shfl_xor(part[v], m, 32);
  }
  if ((lane & 15) == 0) {
    const int rsub = (lane & 16) ? 8 : 0;
    const float ab = a1b[0];
#pragma unroll
    for (int v = 0; v < 8; ++v) outLogit[rowBase + rsub + v] = part[v] + ab;
  }
}

// =====================================================================
// Row kernel: y = mlp_gu([ue[uid]|re[rid]]); wy = y@W^T+b; q = mask*ie[qidx];
// qw = q@W^T+b; logit = att([wy|qw]). 4 waves x 16 rows per WG.
// =====================================================================
__global__ __launch_bounds__(128) void att_rows_kernel(
    const int* __restrict__ pad2, const int* __restrict__ qsrc, int qdiv,
    const float* __restrict__ ue, const float* __restrict__ ie,
    const float* __restrict__ re,
    const bf16_t* __restrict__ arena, int offW, int offAtt0,
    const float* __restrict__ bGu0, const float* __restrict__ bGu1,
    const float* __restrict__ bW, const float* __restrict__ bAtt0,
    const float* __restrict__ att1w, const float* __restrict__ att1b,
    float* __restrict__ yOut, float* __restrict__ wyOut,
    float* __restrict__ logitOut) {
  __shared__ bf16_t sGu0[64 * 136];
  __shared__ bf16_t sGu1[64 * 72];
  __shared__ bf16_t sW[64 * 72];
  __shared__ bf16_t sAtt0[64 * 136];
  __shared__ bf16_t sX[4][16 * 136];
  __shared__ bf16_t sT[4][16 * 72];
  __shared__ bf16_t sY[4][16 * 72];

  const int tid = threadIdx.x, lane = tid & 31, w = tid >> 5;
  // cooperative weight staging (bf16 global -> padded LDS)
  for (int i = tid; i < 64 * 128; i += 128) {
    int r = i >> 7, c = i & 127;
    sGu0[r * 136 + c]  = arena[OFF_GU0 + i];
    sAtt0[r * 136 + c] = arena[offAtt0 + i];
  }
  for (int i = tid; i < 64 * 64; i += 128) {
    int r = i >> 6, c = i & 63;
    sGu1[r * 72 + c] = arena[OFF_GU1 + i];
    sW[r * 72 + c]   = arena[offW + i];
  }
  __syncthreads();

  const int tileBase = blockIdx.x * 64 + w * 16;
  bf16_t* X = sX[w];
  bf16_t* T = sT[w];
  bf16_t* Y = sY[w];

  // gather [ue | re] -> X (float4 loads)
  for (int idx = lane; idx < 16 * 32; idx += 32) {
    int r = idx >> 5, c4 = (idx & 31) * 4;
    int row = tileBase + r;
    int uid = pad2[row * 2], rid = pad2[row * 2 + 1];
    const float* src = (c4 < 64) ? (ue + (size_t)uid * 64 + c4)
                                 : (re + (size_t)rid * 64 + (c4 - 64));
    float4 v = *(const float4*)src;
    bf16_t* d = X + r * 136 + c4;
    d[0] = (bf16_t)v.x; d[1] = (bf16_t)v.y; d[2] = (bf16_t)v.z; d[3] = (bf16_t)v.w;
  }
  __syncthreads();
  wave_gemm64<128>(X, 136, sGu0, 136, bGu0, 1, T, 72, nullptr, 0, lane);   // t1
  __syncthreads();
  wave_gemm64<64>(T, 72, sGu1, 72, bGu1, 0, Y, 72, yOut, tileBase, lane);  // y
  __syncthreads();
  wave_gemm64<64>(Y, 72, sW, 72, bW, 0, X, 136, wyOut, tileBase, lane);    // wy -> X[:,0:64]
  __syncthreads();
  // q = mask * ie[qidx]
  for (int idx = lane; idx < 16 * 16; idx += 32) {
    int r = idx >> 4, c4 = (idx & 15) * 4;
    int row = tileBase + r;
    int uid = pad2[row * 2];
    int qi  = qsrc[row / qdiv];
    float4 v = *(const float4*)(ie + (size_t)qi * 64 + c4);
    float m = (uid > 0) ? 1.f : 0.f;
    bf16_t* d = T + r * 72 + c4;
    d[0] = (bf16_t)(v.x * m); d[1] = (bf16_t)(v.y * m);
    d[2] = (bf16_t)(v.z * m); d[3] = (bf16_t)(v.w * m);
  }
  __syncthreads();
  wave_gemm64<64>(T, 72, sW, 72, bW, 0, X + 64, 136, nullptr, 0, lane);    // qw -> X[:,64:128]
  __syncthreads();
  wave_att_logit(X, 136, sAtt0, 136, bAtt0, att1w, att1b, logitOut, tileBase, lane);
}

// =====================================================================
// kappa rows: logit = att_ff([z_kA@w3^T+b | ie[iid]@w3^T+b])
// =====================================================================
__global__ __launch_bounds__(128) void kappa_rows_kernel(
    const float* __restrict__ zkA, const int* __restrict__ iip,
    const float* __restrict__ ie, const bf16_t* __restrict__ arena,
    const float* __restrict__ bW3, const float* __restrict__ bAtt0,
    const float* __restrict__ att1w, const float* __restrict__ att1b,
    float* __restrict__ logitOut) {
  __shared__ bf16_t sW[64 * 72];
  __shared__ bf16_t sAtt0[64 * 136];
  __shared__ bf16_t sX[4][16 * 136];
  __shared__ bf16_t sT[4][16 * 72];

  const int tid = threadIdx.x, lane = tid & 31, w = tid >> 5;
  for (int i = tid; i < 64 * 128; i += 128) {
    int r = i >> 7, c = i & 127;
    sAtt0[r * 136 + c] = arena[OFF_ATTFF0 + i];
  }
  for (int i = tid; i < 64 * 64; i += 128) {
    int r = i >> 6, c = i & 63;
    sW[r * 72 + c] = arena[OFF_W3 + i];
  }
  __syncthreads();

  const int tileBase = blockIdx.x * 64 + w * 16;
  bf16_t* X = sX[w];
  bf16_t* T = sT[w];

  for (int idx = lane; idx < 16 * 16; idx += 32) {          // z_kA tile
    int r = idx >> 4, c4 = (idx & 15) * 4;
    float4 v = *(const float4*)(zkA + (size_t)(tileBase + r) * 64 + c4);
    bf16_t* d = T + r * 72 + c4;
    d[0] = (bf16_t)v.x; d[1] = (bf16_t)v.y; d[2] = (bf16_t)v.z; d[3] = (bf16_t)v.w;
  }
  __syncthreads();
  wave_gemm64<64>(T, 72, sW, 72, bW3, 0, X, 136, nullptr, 0, lane);
  __syncthreads();
  for (int idx = lane; idx < 16 * 16; idx += 32) {          // ie tile (no mask here)
    int r = idx >> 4, c4 = (idx & 15) * 4;
    int qi = iip[tileBase + r];
    float4 v = *(const float4*)(ie + (size_t)qi * 64 + c4);
    bf16_t* d = T + r * 72 + c4;
    d[0] = (bf16_t)v.x; d[1] = (bf16_t)v.y; d[2] = (bf16_t)v.z; d[3] = (bf16_t)v.w;
  }
  __syncthreads();
  wave_gemm64<64>(T, 72, sW, 72, bW3, 0, X + 64, 136, nullptr, 0, lane);
  __syncthreads();
  wave_att_logit(X, 136, sAtt0, 136, bAtt0, att1w, att1b, logitOut, tileBase, lane);
}

// =====================================================================
// One LSTM layer over T steps, 16 batch rows/WG, gates via WMMA.
// =====================================================================
__device__ __forceinline__ float sigmf(float x) { return 1.f / (1.f + expf(-x)); }

__global__ __launch_bounds__(128) void lstm_layer_kernel(
    const float* __restrict__ xseq, float* __restrict__ hseq,
    const float* __restrict__ h0, const float* __restrict__ c0,
    int l, int h0div, int T, const bf16_t* __restrict__ lw,
    const float* __restrict__ biasComb, float* __restrict__ hOut) {
  __shared__ bf16_t wih[256 * 68];
  __shared__ bf16_t whh[256 * 68];
  __shared__ bf16_t xbuf[16 * 68];
  __shared__ bf16_t hbuf[16 * 68];
  __shared__ float  cbuf[16][64];
  __shared__ float  gbuf[16][268];
  __shared__ float  sbias[256];

  const int tid = threadIdx.x, lane = tid & 31, w = tid >> 5;
  const int rbase = blockIdx.x * 16;
  const bf16_t* Wi = lw + (size_t)l * 32768;
  const bf16_t* Wh = Wi + 16384;
  for (int i = tid; i < 256 * 64; i += 128) {
    int r = i >> 6, c = i & 63;
    wih[r * 68 + c] = Wi[i];
    whh[r * 68 + c] = Wh[i];
  }
  for (int i = tid; i < 256; i += 128) sbias[i] = biasComb[l * 256 + i];
  for (int i = tid; i < 16 * 64; i += 128) {
    int r = i >> 6, d = i & 63;
    int hr = (rbase + r) / h0div;
    hbuf[r * 68 + d] = (bf16_t)h0[((size_t)l * 256 + hr) * 64 + d];
    cbuf[r][d]       = c0[((size_t)l * 256 + hr) * 64 + d];
  }
  __syncthreads();

  const int nsub = lane & 15;
  const int rsub = (lane & 16) ? 8 : 0;
  for (int t = 0; t < T; ++t) {
    // stage x_t
    for (int i = tid; i < 256; i += 128) {
      int r = i >> 4, c4 = (i & 15) * 4;
      float4 v = *(const float4*)(xseq + ((size_t)(rbase + r) * T + t) * 64 + c4);
      bf16_t* d = xbuf + r * 68 + c4;
      d[0] = (bf16_t)v.x; d[1] = (bf16_t)v.y; d[2] = (bf16_t)v.z; d[3] = (bf16_t)v.w;
    }
    __syncthreads();
    bf16x16 ax0 = pack_frag(xbuf, 68, 0, lane);
    bf16x16 ax1 = pack_frag(xbuf, 68, 32, lane);
    bf16x16 ah0 = pack_frag(hbuf, 68, 0, lane);
    bf16x16 ah1 = pack_frag(hbuf, 68, 32, lane);
#pragma unroll
    for (int nt = 0; nt < 4; ++nt) {
      const int n0 = w * 64 + nt * 16;
      floatx8 acc = zero8();
      acc = wmma_bf16(ax0, pack_frag(wih + n0 * 68, 68, 0, lane), acc);
      acc = wmma_bf16(ax1, pack_frag(wih + n0 * 68, 68, 32, lane), acc);
      acc = wmma_bf16(ah0, pack_frag(whh + n0 * 68, 68, 0, lane), acc);
      acc = wmma_bf16(ah1, pack_frag(whh + n0 * 68, 68, 32, lane), acc);
      const int n = n0 + nsub;
      const float bb = sbias[n];
#pragma unroll
      for (int v = 0; v < 8; ++v) gbuf[v + rsub][n] = acc[v] + bb;
    }
    __syncthreads();
#pragma unroll
    for (int j = 0; j < 8; ++j) {
      const int e = tid * 8 + j;
      const int r = e >> 6, d = e & 63;
      const float gi = gbuf[r][d],       gf = gbuf[r][64 + d];
      const float gg = gbuf[r][128 + d], go = gbuf[r][192 + d];
      const float c = sigmf(gf) * cbuf[r][d] + sigmf(gi) * tanhf(gg);
      const float h = sigmf(go) * tanhf(c);
      cbuf[r][d] = c;
      hbuf[r * 68 + d] = (bf16_t)h;
      hseq[((size_t)(rbase + r) * T + t) * 64 + d] = h;
      if (hOut != nullptr && t == T - 1)
        hOut[(size_t)(rbase + r) * 64 + d] = h;
    }
    __syncthreads();
  }
}

// =====================================================================
// Masked softmax over nU + weighted sum + aggre linear (relu).
// =====================================================================
__global__ __launch_bounds__(64) void att_reduce_kernel(
    const float* __restrict__ logits, const int* __restrict__ midx, int mStride,
    const float* __restrict__ vals, const float* __restrict__ W,
    const float* __restrict__ bias, float* __restrict__ out, int nU) {
  __shared__ float wgt[64];
  __shared__ float red[64];
  __shared__ float s[64];
  const int b = blockIdx.x, t = threadIdx.x;
  float wv = 0.f;
  if (t < nU) {
    const int row = b * nU + t;
    const float m = (midx[(size_t)row * mStride] > 0) ? 1.f : 0.f;
    wv = expf(logits[row]) * m;
  }
  wgt[t] = wv;
  red[t] = wv;
  __syncthreads();
  for (int s2 = 32; s2 > 0; s2 >>= 1) {
    if (t < s2) red[t] += red[t + s2];
    __syncthreads();
  }
  const float denom = red[0] + 1e-10f;
  float acc = 0.f;
  for (int u = 0; u < nU; ++u)
    acc += wgt[u] * vals[(size_t)(b * nU + u) * 64 + t];
  s[t] = acc / denom;
  __syncthreads();
  float o = bias[t];
  for (int j = 0; j < 64; ++j) o += W[t * 64 + j] * s[j];
  out[(size_t)b * 64 + t] = fmaxf(o, 0.f);
}

// =====================================================================
// small utility kernels
// =====================================================================
__global__ void cvt_bf16_kernel(const float* __restrict__ src,
                                bf16_t* __restrict__ dst, int n) {
  int i = blockIdx.x * 256 + threadIdx.x;
  if (i < n) dst[i] = (bf16_t)src[i];
}

__global__ void addb_kernel(const float* __restrict__ a, const float* __restrict__ b,
                            float* __restrict__ o, int n) {
  int i = blockIdx.x * 256 + threadIdx.x;
  if (i < n) o[i] = a[i] + b[i];
}

__global__ void mul_kernel(const float* __restrict__ a, const float* __restrict__ b,
                           float* __restrict__ o, int n) {
  int i = blockIdx.x * 256 + threadIdx.x;
  if (i < n) o[i] = a[i] * b[i];
}

__global__ __launch_bounds__(128) void combine_kernel(
    const float* __restrict__ zjL, const float* __restrict__ zjS,
    const float* __restrict__ zjN,
    const float* __restrict__ C1w, const float* __restrict__ C1b,
    const float* __restrict__ C2w, const float* __restrict__ C2b,
    float* __restrict__ out) {
  __shared__ float x[128];
  __shared__ float h[128];
  const int b = blockIdx.x, t = threadIdx.x;
  x[t] = (t < 64) ? (zjL[b * 64 + t] * zjS[b * 64 + t]) : zjN[b * 64 + (t - 64)];
  __syncthreads();
  float a = C1b[t];
  for (int j = 0; j < 128; ++j) a += C1w[t * 128 + j] * x[j];
  h[t] = fmaxf(a, 0.f);
  __syncthreads();
  if (t < 64) {
    float o = C2b[t];
    for (int j = 0; j < 128; ++j) o += C2w[t * 128 + j] * h[j];
    out[(size_t)b * 64 + t] = fmaxf(o, 0.f);
  }
}

// =====================================================================
extern "C" void kernel_launch(void* const* d_in, const int* in_sizes, int n_in,
                              void* d_out, int out_size, void* d_ws, size_t ws_size,
                              hipStream_t stream) {
  (void)in_sizes; (void)n_in; (void)out_size; (void)ws_size;
  // inputs in setup_inputs() dict order, params flattened in insertion order
  const int*   iids = (const int*)d_in[0];
  const int*   iup  = (const int*)d_in[1];   // i_user_pad [B][U][2]
  const int*   iip  = (const int*)d_in[2];   // i_item_pad [B][K]
  const int*   iiup = (const int*)d_in[3];   // i_item_user_pad [B][K][U2][2]
  const float* ue   = (const float*)d_in[4];
  const float* ie   = (const float*)d_in[5];
  const float* re   = (const float*)d_in[6];
  const float *w1w = (const float*)d_in[7],  *w1b = (const float*)d_in[8];
  const float *w2w = (const float*)d_in[9],  *w2b = (const float*)d_in[10];
  const float *w3w = (const float*)d_in[11], *w3b = (const float*)d_in[12];
  const float *gu0w = (const float*)d_in[13], *gu0b = (const float*)d_in[14];
  const float *gu1w = (const float*)d_in[15], *gu1b = (const float*)d_in[16];
  const float *aiu0w = (const float*)d_in[17], *aiu0b = (const float*)d_in[18];
  const float *aiu1w = (const float*)d_in[19], *aiu1b = (const float*)d_in[20];
  const float *agiuw = (const float*)d_in[21], *agiub = (const float*)d_in[22];
  const float *aif0w = (const float*)d_in[23], *aif0b = (const float*)d_in[24];
  const float *aif1w = (const float*)d_in[25], *aif1b = (const float*)d_in[26];
  const float *agifw = (const float*)d_in[27], *agifb = (const float*)d_in[28];
  const float *aff0w = (const float*)d_in[29], *aff0b = (const float*)d_in[30];
  const float *aff1w = (const float*)d_in[31], *aff1b = (const float*)d_in[32];
  const float *agffw = (const float*)d_in[33], *agffb = (const float*)d_in[34];
  // lstm layer l: d_in[35+4l]=Wih, [36+4l]=Whh, [37+4l]=bih, [38+4l]=bhh
  const float *c1w = (const float*)d_in[51], *c1b = (const float*)d_in[52];
  const float *c2w = (const float*)d_in[53], *c2b = (const float*)d_in[54];
  const float *h0 = (const float*)d_in[55], *c0 = (const float*)d_in[56];
  float* out = (float*)d_out;

  bf16_t* arena = (bf16_t*)d_ws;
  float*  F     = (float*)((char*)d_ws + ARENA_BYTES);
  float *biasc = F + F_BIASC, *y1 = F + F_Y1, *w1y = F + F_W1Y, *log1 = F + F_LOG1;
  float *zjL = F + F_ZJL, *zjS = F + F_ZJS, *lseq1 = F + F_LSEQ1;
  float *ys = F + F_YS, *log2 = F + F_LOG2, *zkL = F + F_ZKL, *zkS = F + F_ZKS;
  float *zkA = F + F_ZKA, *klog = F + F_KLOG, *zjN = F + F_ZJN, *lseq2 = F + F_LSEQ2;

  auto cvt = [&](const float* s, int off, int n) {
    cvt_bf16_kernel<<<(n + 255) / 256, 256, 0, stream>>>(s, arena + off, n);
  };
  cvt(gu0w, OFF_GU0, 8192);
  cvt(gu1w, OFF_GU1, 4096);
  cvt(w1w, OFF_W1, 4096);
  cvt(w2w, OFF_W2, 4096);
  cvt(w3w, OFF_W3, 4096);
  cvt(aiu0w, OFF_ATTIU0, 8192);
  cvt(aif0w, OFF_ATTIF0, 8192);
  cvt(aff0w, OFF_ATTFF0, 8192);
  for (int l = 0; l < 4; ++l) {
    cvt((const float*)d_in[35 + l * 4], OFF_LSTM + l * 32768, 16384);
    cvt((const float*)d_in[36 + l * 4], OFF_LSTM + l * 32768 + 16384, 16384);
    addb_kernel<<<1, 256, 0, stream>>>((const float*)d_in[37 + l * 4],
                                       (const float*)d_in[38 + l * 4],
                                       biasc + l * 256, 256);
  }

  // ---- branch 1: item-user ----
  att_rows_kernel<<<200, 128, 0, stream>>>(iup, iids, U_, ue, ie, re, arena,
      OFF_W1, OFF_ATTIU0, gu0b, gu1b, w1b, aiu0b, aiu1w, aiu1b, y1, w1y, log1);
  att_reduce_kernel<<<256, 64, 0, stream>>>(log1, iup, 2, w1y, agiuw, agiub, zjL, U_);
  for (int l = 0; l < 4; ++l) {
    const float* in = (l % 2 == 0) ? y1 : lseq1;
    float*       o  = (l % 2 == 0) ? lseq1 : y1;
    lstm_layer_kernel<<<16, 128, 0, stream>>>(in, o, h0, c0, l, 1, U_,
        arena + OFF_LSTM, biasc, (l == 3) ? zjS : nullptr);
  }

  // ---- branch 2: item-item (social) ----
  att_rows_kernel<<<6000, 128, 0, stream>>>(iiup, iip, U2_, ue, ie, re, arena,
      OFF_W2, OFF_ATTIF0, gu0b, gu1b, w2b, aif0b, aif1w, aif1b, ys, nullptr, log2);
  att_reduce_kernel<<<12800, 64, 0, stream>>>(log2, iiup, 2, ys, agifw, agifb, zkL, U2_);
  for (int l = 0; l < 4; ++l) {
    const float* in = (l % 2 == 0) ? ys : lseq2;
    float*       o  = (l % 2 == 0) ? lseq2 : ys;
    lstm_layer_kernel<<<800, 128, 0, stream>>>(in, o, h0, c0, l, K_, U2_,
        arena + OFF_LSTM, biasc, (l == 3) ? zkS : nullptr);
  }
  mul_kernel<<<(12800 * 64 + 255) / 256, 256, 0, stream>>>(zkL, zkS, zkA, 12800 * 64);

  // ---- item-level attention + combine ----
  kappa_rows_kernel<<<200, 128, 0, stream>>>(zkA, iip, ie, arena, w3b, aff0b,
                                             aff1w, aff1b, klog);
  att_reduce_kernel<<<256, 64, 0, stream>>>(klog, iip, 1, zkA, agffw, agffb, zjN, K_);
  combine_kernel<<<256, 128, 0, stream>>>(zjL, zjS, zjN, c1w, c1b, c2w, c2b, out);
}
